// MultiHeadAttention_59227599012491
// MI455X (gfx1250) — matrix operand order repaired and checked by reading, not compiled, not run on hardware
//
#include <hip/hip_runtime.h>
#include <math.h>

// ---------------------------------------------------------------------------
// Problem constants (B=4, S=4096, H*W = D = 256)
// ---------------------------------------------------------------------------
#define DMODEL 256
#define SEQ    4096
#define BATCH  4
#define NROWS  (BATCH * SEQ)   // 16384
#define NCHUNK (SEQ / 32)      // 128 key chunks of 32

typedef __attribute__((ext_vector_type(16))) __bf16        v16bf;
typedef __attribute__((ext_vector_type(8)))  __bf16        v8bf;
typedef __attribute__((ext_vector_type(8)))  float         v8f;
typedef __attribute__((ext_vector_type(4)))  unsigned int  v4u;
typedef __attribute__((ext_vector_type(8)))  unsigned int  v8u;

union F16x16 { v16bf v; v8bf h[2]; };

__device__ __forceinline__ v16bf ld_frag_pair(const __bf16* p0, const __bf16* p1) {
  F16x16 u;
  u.h[0] = *reinterpret_cast<const v8bf*>(p0);
  u.h[1] = *reinterpret_cast<const v8bf*>(p1);
  return u.v;
}

__device__ __forceinline__ v8f wmma_bf16(v16bf a, v16bf b, v8f c) {
  return __builtin_amdgcn_wmma_f32_16x16x32_bf16(
      /*neg_a=*/false, a, /*neg_b=*/false, b,
      /*c_mod=*/(short)0, c, /*reuse_a=*/false, /*reuse_b=*/false);
}

__device__ __forceinline__ void sched_fence() {
#if __has_builtin(__builtin_amdgcn_sched_barrier)
  __builtin_amdgcn_sched_barrier(0);   // keep load phase / compute phase apart
#endif
}

// ---------------------------------------------------------------------------
// f32 -> bf16 elementwise conversion (x and weight matrices)
// ---------------------------------------------------------------------------
__global__ void cvt_f32_bf16(const float* __restrict__ src,
                             __bf16* __restrict__ dst, int n) {
  int i = blockIdx.x * blockDim.x + threadIdx.x;
  int stride = gridDim.x * blockDim.x;
  for (; i < n; i += stride) dst[i] = (__bf16)src[i];
}

// ---------------------------------------------------------------------------
// Projection GEMM:  Y[m,n] = act( sum_k X[m,k] * W[n,k] + bias[n] )
// All 8 B fragments are fetched in one clause (sched_barrier keeps them
// clustered -> single s_wait_loadcnt), then 8 WMMAs issue back-to-back.
// ---------------------------------------------------------------------------
__global__ __launch_bounds__(256)
void proj_gemm(const __bf16* __restrict__ X, const __bf16* __restrict__ Wb,
               const float* __restrict__ bias,
               __bf16* __restrict__ Ybf, float* __restrict__ Yf32,
               int softplus_act) {
  const int lane = threadIdx.x & 31;
  const int wave = threadIdx.x >> 5;
  const int row0 = blockIdx.x * 128 + wave * 16;
  const int m  = lane & 15;   // A-matrix row / C-matrix column index
  const int hi = lane >> 4;   // upper-half-lane flag

  // ----- A fragments: rows of X, K = 0..255 in 8 fragments of K=32 -----
  const __bf16* xrow = X + (size_t)(row0 + m) * DMODEL;
  v16bf aq[8];
#pragma unroll
  for (int i = 0; i < 8; ++i) {
    const int klo = hi * 16;  // K order used here (B matches): contiguous {0..15} / {16..31}
    aq[i] = ld_frag_pair(xrow + 32 * i + klo, xrow + 32 * i + klo + 8);
  }

  // ----- 16 column tiles of 16 outputs each -----
  for (int t = 0; t < 16; ++t) {
    const int ncol = 16 * t + m;                  // B-layout: lane -> N = lane%16
    const __bf16* wrow = Wb + (size_t)ncol * DMODEL + hi * 16;  // K contiguous

    v16bf bf[8];
#pragma unroll
    for (int i = 0; i < 8; ++i)
      bf[i] = ld_frag_pair(wrow + 32 * i, wrow + 32 * i + 8);
    sched_fence();  // do not sink loads into the WMMA chain

    v8f acc = {0.f, 0.f, 0.f, 0.f, 0.f, 0.f, 0.f, 0.f};
#pragma unroll
    for (int i = 0; i < 8; ++i)
      acc = wmma_bf16(aq[i], bf[i], acc);

    const float bn = bias[ncol];
#pragma unroll
    for (int g = 0; g < 8; ++g) {
      float v = acc[g] + bn;
      if (softplus_act) {
        // stable softplus: max(x,0) + log1p(exp(-|x|))
        v = fmaxf(v, 0.f) + log1pf(__expf(-fabsf(v)));
      }
      const int row = row0 + g + 8 * hi;          // C-layout: M = g + 8*(lane>>4)
      const size_t o = (size_t)row * DMODEL + 16 * t + m;
      if (Yf32) Yf32[o] = v;
      else      Ybf[o] = (__bf16)v;
    }
  }
}

// ---------------------------------------------------------------------------
// Fused attention without softmax:  O = (Q K^T) V   per batch.
// Workgroup: 128 Q rows (8 waves x 16 rows), 32-key chunks, DOUBLE-BUFFERED:
//   Kch[2][32][264] : K chunk staged by the TENSOR DATA MOVER; the TDM for
//                     chunk i+1 is issued before compute on chunk i and only
//                     waited on (s_wait_tensorcnt) after the WMMAs -> the DMA
//                     runs entirely under the compute shadow.
//   Vt [2][256][40] : V chunk transposed; staged for i+1 during compute on i.
//   Ssc[8][16][40]  : per-wave score-tile C-layout -> A-layout round-trip.
// Static LDS ~85 KB: legal on CDNA5 (320 KB per workgroup).
// ---------------------------------------------------------------------------
__global__ __launch_bounds__(256)
void attn_kernel(const __bf16* __restrict__ Q, const __bf16* __restrict__ K,
                 const __bf16* __restrict__ V, __bf16* __restrict__ O) {
  __shared__ __bf16 Kch[2][32][264];
  __shared__ __bf16 Vt[2][256][40];
  __shared__ __bf16 Ssc[8][16][40];

  const int tid  = threadIdx.x;
  const int lane = tid & 31;
  const int wave = tid >> 5;
  const int b    = blockIdx.y;
  const int row0 = blockIdx.x * 128 + wave * 16;
  const int m  = lane & 15;
  const int hi = lane >> 4;

  const __bf16* Qb = Q + (size_t)b * SEQ * DMODEL;
  const __bf16* Kb = K + (size_t)b * SEQ * DMODEL;
  const __bf16* Vb = V + (size_t)b * SEQ * DMODEL;

  // Resident A fragments for this wave's 16 Q rows (full K=256).
  const __bf16* qrow = Qb + (size_t)(row0 + m) * DMODEL;
  v16bf aq[8];
#pragma unroll
  for (int i = 0; i < 8; ++i) {
    const int klo = hi * 16;
    aq[i] = ld_frag_pair(qrow + 32 * i + klo, qrow + 32 * i + klo + 8);
  }

  // O accumulator: 16 column tiles x (16x16 f32) = full 16x256 slice.
  v8f acc[16];
#pragma unroll
  for (int n = 0; n < 16; ++n)
    acc[n] = (v8f){0.f, 0.f, 0.f, 0.f, 0.f, 0.f, 0.f, 0.f};

  // ---- helpers --------------------------------------------------------
  // Issue one TDM descriptor copying 32x256 bf16 rows into Kch[buf] with the
  // 16B-per-512B LDS padding applied by the DMA itself.
  auto tdm_stage_k = [&](int kb0, int buf) {
    const unsigned lds = (unsigned)(uintptr_t)&Kch[buf][0][0];
    const unsigned long long ga =
        (unsigned long long)(uintptr_t)(Kb + (size_t)kb0 * DMODEL);
    v4u g0;
    g0.x = 1u;                                   // count = 1 valid descriptor
    g0.y = lds;                                  // LDS destination address
    g0.z = (unsigned)ga;                         // global_addr[31:0]
    g0.w = (unsigned)((ga >> 32) & 0x01FFFFFFu)  // global_addr[56:32]
           | (2u << 30);                         // type = 2 ("image")
    v8u g1;
    g1.s0 = (1u << 16)      // data_size = 1 (2 bytes)
          | (1u << 20)      // pad_enable
          | (6u << 22)      // pad_interval: 128 DWORDs (=512B)
          | (3u << 25);     // pad_amount: 4 DWORDs (=16B)
    g1.s1 = (256u << 16);   // tensor_dim0 = 256
    g1.s2 = (32u  << 16);   // tensor_dim1 = 32
    g1.s3 = (256u << 16);   // tile_dim0 = 256
    g1.s4 = 32u;            // tile_dim1 = 32
    g1.s5 = 256u;           // tensor_dim0_stride = 256
    g1.s6 = 0u;
    g1.s7 = 0u;
    asm volatile("tensor_load_to_lds %0, %1" :: "s"(g0), "s"(g1) : "memory");
  };

  auto stage_v = [&](int kb0, int buf) {
    const int keyb = tid >> 5;          // 0..7
    const int d0   = (tid & 31) * 8;    // 0..248
#pragma unroll
    for (int j = 0; j < 4; ++j) {
      const int key = keyb + 8 * j;
      v8bf vv = *reinterpret_cast<const v8bf*>(Vb + (size_t)(kb0 + key) * DMODEL + d0);
#pragma unroll
      for (int e = 0; e < 8; ++e) Vt[buf][d0 + e][key] = vv[e];
    }
  };
  // ---------------------------------------------------------------------

  // Prologue: stage chunk 0 into buffer 0.
  if (wave == 0) {
    tdm_stage_k(0, 0);
    __builtin_amdgcn_s_wait_tensorcnt(0);
  }
  stage_v(0, 0);
  __syncthreads();

  for (int i = 0; i < NCHUNK; ++i) {
    const int cur = i & 1;
    const int nxt = cur ^ 1;

    // Kick the async pipeline for chunk i+1 before computing on chunk i.
    if (i + 1 < NCHUNK) {
      if (wave == 0) tdm_stage_k((i + 1) * 32, nxt);   // async DMA, no wait yet
      stage_v((i + 1) * 32, nxt);                      // overlaps the WMMAs below
      if (i + 2 < NCHUNK)                              // cache prefetch one ahead
        __builtin_prefetch(Vb + (size_t)(i + 2) * 32 * DMODEL + (size_t)tid * 32, 0, 1);
    }

    // ---- scores: S(16x32) = Q_slice x Kchunk^T, two 16-key tiles ----
#pragma unroll
    for (int t = 0; t < 2; ++t) {
      const __bf16* krow = &Kch[cur][16 * t + m][hi * 16];  // B: N=key, K=d
      v8f s = {0.f, 0.f, 0.f, 0.f, 0.f, 0.f, 0.f, 0.f};
      v16bf kcur = ld_frag_pair(krow, krow + 8);
#pragma unroll
      for (int k = 0; k < 8; ++k) {
        v16bf knext = kcur;
        if (k < 7) {  // double buffer the ds loads under the WMMA
          const __bf16* kp = krow + 32 * (k + 1);
          knext = ld_frag_pair(kp, kp + 8);
        }
        s = wmma_bf16(aq[k], kcur, s);
        kcur = knext;
      }
      // C-layout -> row-major bf16 scratch (per-wave private region)
#pragma unroll
      for (int g = 0; g < 8; ++g)
        Ssc[wave][g + 8 * hi][16 * t + m] = (__bf16)s[g];
    }
    // Cross-lane LDS RAW inside the wave: wait for our DS stores to land.
    asm volatile("s_wait_dscnt 0x0" ::: "memory");

    // Re-read the 16x32 score tile in A-layout.
    const __bf16* srow = &Ssc[wave][m][0];
    const int klo = hi * 16;
    v16bf sfrag = ld_frag_pair(srow + klo, srow + klo + 8);

    // ---- O += S(16x32) x Vchunk(32x256), 16 column tiles ----
    {
      const __bf16* vp0 = &Vt[cur][m][hi * 16];
      v16bf vcur = ld_frag_pair(vp0, vp0 + 8);
#pragma unroll
      for (int n = 0; n < 16; ++n) {
        v16bf vnext = vcur;
        if (n < 15) {
          const __bf16* vp = &Vt[cur][16 * (n + 1) + m][hi * 16];
          vnext = ld_frag_pair(vp, vp + 8);
        }
        acc[n] = wmma_bf16(sfrag, vcur, acc[n]);
        vcur = vnext;
      }
    }

    // Drain the DMA for chunk i+1, then publish both staged buffers.
    if (i + 1 < NCHUNK && wave == 0)
      __builtin_amdgcn_s_wait_tensorcnt(0);
    __syncthreads();
  }

  // ---- store O slice as bf16 for the output projection ----
  __bf16* orow = O + ((size_t)b * SEQ + row0) * DMODEL;
#pragma unroll
  for (int n = 0; n < 16; ++n) {
#pragma unroll
    for (int g = 0; g < 8; ++g)
      orow[(size_t)(g + 8 * hi) * DMODEL + 16 * n + m] = (__bf16)acc[n][g];
  }
}

// ---------------------------------------------------------------------------
// Launcher. Workspace: 5 row buffers (8 MB each) + 4 bf16 weights (128 KB).
// ---------------------------------------------------------------------------
extern "C" void kernel_launch(void* const* d_in, const int* in_sizes, int n_in,
                              void* d_out, int out_size, void* d_ws, size_t ws_size,
                              hipStream_t stream) {
  const float* x  = (const float*)d_in[0];
  const float* Wq = (const float*)d_in[1];
  const float* bq = (const float*)d_in[2];
  const float* Wk = (const float*)d_in[3];
  const float* bk = (const float*)d_in[4];
  const float* Wv = (const float*)d_in[5];
  const float* bv = (const float*)d_in[6];
  const float* Wo = (const float*)d_in[7];
  const float* bo = (const float*)d_in[8];
  float* out = (float*)d_out;

  char* ws = (char*)d_ws;
  size_t off = 0;
  auto carve = [&](size_t bytes) -> void* {
    void* p = ws + off;
    off += (bytes + 255) & ~(size_t)255;
    return p;
  };
  __bf16* xb  = (__bf16*)carve((size_t)NROWS * DMODEL * 2);
  __bf16* qb  = (__bf16*)carve((size_t)NROWS * DMODEL * 2);
  __bf16* kb  = (__bf16*)carve((size_t)NROWS * DMODEL * 2);
  __bf16* vb  = (__bf16*)carve((size_t)NROWS * DMODEL * 2);
  __bf16* rb  = (__bf16*)carve((size_t)NROWS * DMODEL * 2);
  __bf16* wqb = (__bf16*)carve((size_t)DMODEL * DMODEL * 2);
  __bf16* wkb = (__bf16*)carve((size_t)DMODEL * DMODEL * 2);
  __bf16* wvb = (__bf16*)carve((size_t)DMODEL * DMODEL * 2);
  __bf16* wob = (__bf16*)carve((size_t)DMODEL * DMODEL * 2);
  (void)ws_size; (void)in_sizes; (void)n_in; (void)out_size;

  // 1) f32 -> bf16 conversions
  cvt_f32_bf16<<<512, 256, 0, stream>>>(x,  xb,  NROWS * DMODEL);
  cvt_f32_bf16<<<64,  256, 0, stream>>>(Wq, wqb, DMODEL * DMODEL);
  cvt_f32_bf16<<<64,  256, 0, stream>>>(Wk, wkb, DMODEL * DMODEL);
  cvt_f32_bf16<<<64,  256, 0, stream>>>(Wv, wvb, DMODEL * DMODEL);
  cvt_f32_bf16<<<64,  256, 0, stream>>>(Wo, wob, DMODEL * DMODEL);

  // 2) Q/K/V projections (softplus on q,k; none on v)
  proj_gemm<<<NROWS / 128, 256, 0, stream>>>(xb, wqb, bq, qb, nullptr, 1);
  proj_gemm<<<NROWS / 128, 256, 0, stream>>>(xb, wkb, bk, kb, nullptr, 1);
  proj_gemm<<<NROWS / 128, 256, 0, stream>>>(xb, wvb, bv, vb, nullptr, 0);

  // 3) fused (QK^T)V, no softmax
  attn_kernel<<<dim3(SEQ / 128, BATCH), 256, 0, stream>>>(qb, kb, vb, rb);

  // 4) output projection -> f32 result
  proj_gemm<<<NROWS / 128, 256, 0, stream>>>(rb, wob, bo, nullptr, out, 0);
}
